// GQAAttention_82051055222889
// MI455X (gfx1250) — matrix operand order repaired and checked by reading, not compiled, not run on hardware
//
#include <hip/hip_runtime.h>
#include <hip/hip_bf16.h>

// ---------------- problem constants ----------------
#define SEQ   2048
#define HID   2048
#define NH    32
#define NKV   8
#define HD    64
#define NQKV  3072          // NH*HD + NKV*HD + NKV*HD = 2048+512+512
#define GROUPS (NH / NKV)   // 4

typedef _Float16 v16h __attribute__((ext_vector_type(16)));
typedef _Float16 v8h  __attribute__((ext_vector_type(8)));
typedef float    v8f  __attribute__((ext_vector_type(8)));

union V16 { v16h v; v8h h[2]; };

static __device__ __forceinline__ v8f zero8() {
    v8f z = {0.f,0.f,0.f,0.f,0.f,0.f,0.f,0.f};
    return z;
}

static __device__ __forceinline__ v8f wmma_f16(v16h a, v16h b, v8f c) {
    // D = A(16x32 f16) x B(32x16 f16) + C(16x16 f32)   -> v_wmma_f32_16x16x32_f16
    return __builtin_amdgcn_wmma_f32_16x16x32_f16(
        /*neg_a=*/false, a, /*neg_b=*/false, b,
        /*c_mod=*/(short)0, c, /*reuse_a=*/false, /*reuse_b=*/false);
}

// ---------------------------------------------------------------------------
// Kernel 0a: wq|wk|wv (f32, [K][N]) -> wqkvT (f16, [N=3072][K=2048])
// ---------------------------------------------------------------------------
__global__ __launch_bounds__(256) void prep_wqkvT(
    const float* __restrict__ wq, const float* __restrict__ wk,
    const float* __restrict__ wv, _Float16* __restrict__ wqkvT)
{
    size_t tid = (size_t)blockIdx.x * 256 + threadIdx.x;   // [0, 3072*2048)
    int n = (int)(tid / HID);
    int k = (int)(tid % HID);
    float v;
    if (n < NH * HD)               v = wq[(size_t)k * (NH * HD)  + n];
    else if (n < NH * HD + NKV*HD) v = wk[(size_t)k * (NKV * HD) + (n - NH * HD)];
    else                           v = wv[(size_t)k * (NKV * HD) + (n - NH * HD - NKV * HD)];
    wqkvT[(size_t)n * HID + k] = (_Float16)v;
}

// ---------------------------------------------------------------------------
// Kernel 0b: wo (f32, [K][N]) -> woT (f16, [N=2048][K=2048])
// ---------------------------------------------------------------------------
__global__ __launch_bounds__(256) void prep_woT(
    const float* __restrict__ wo, _Float16* __restrict__ woT)
{
    size_t tid = (size_t)blockIdx.x * 256 + threadIdx.x;   // [0, 2048*2048)
    int n = (int)(tid / HID);
    int k = (int)(tid % HID);
    woT[(size_t)n * HID + k] = (_Float16)wo[(size_t)k * HID + n];
}

// ---------------------------------------------------------------------------
// Kernel 1: qkv = x @ [wq|wk|wv]   (f32 x, f16 weights, f32 accum, f16 out)
// Register-blocked: one wave computes a 32x64 macro-tile = 2(M) x 4(N) WMMA
// tiles; each A fragment reused 4x, each B fragment 2x -> 8 wmma / K-step.
//
// Fragment layouts used here (16-bit, wave32), lane m=lane&15, hi=lane>>4:
//   A 16x32: halves 0..7 -> K = hi*16 + j ; halves 8..15 -> K = hi*16 + 8 + j
//   B 32x16: lane holds column n; halves j -> K = hi*16 + j  (contiguous)
//   C 16x16: VGPR r -> row (hi ? 8+r : r), col = lane&15
// ---------------------------------------------------------------------------
#define QKV_NTW (NQKV / 64)   // 48 macro-tiles along N
__global__ __launch_bounds__(128) void gemm_qkv(
    const float* __restrict__ x, const _Float16* __restrict__ wqkvT,
    _Float16* __restrict__ qkv)
{
    const int lane = threadIdx.x & 31;
    const int w    = threadIdx.x >> 5;
    const int tile = blockIdx.x * 4 + w;          // 64*48 = 3072 macro-tiles
    const int mt   = tile / QKV_NTW;
    const int nt   = tile % QKV_NTW;
    const int m0   = mt * 32, n0 = nt * 64;
    const int mrow = lane & 15, hi = lane >> 4;

    v8f acc[2][4];
#pragma unroll
    for (int mi = 0; mi < 2; ++mi)
#pragma unroll
        for (int ni = 0; ni < 4; ++ni) acc[mi][ni] = zero8();

    const float* xr[2];
    xr[0] = x + (size_t)(m0 + mrow) * HID;
    xr[1] = x + (size_t)(m0 + 16 + mrow) * HID;

    for (int k0 = 0; k0 < HID; k0 += 32) {
        __builtin_prefetch(xr[0] + k0 + 64, 0, 0);
        V16 a[2];
#pragma unroll
        for (int mi = 0; mi < 2; ++mi) {
#pragma unroll
            for (int j = 0; j < 8; ++j) a[mi].v[j]     = (_Float16)xr[mi][k0 + hi * 16 + j];
#pragma unroll
            for (int j = 0; j < 8; ++j) a[mi].v[8 + j] = (_Float16)xr[mi][k0 + hi * 16 + 8 + j];
        }
#pragma unroll
        for (int ni = 0; ni < 4; ++ni) {
            v16h b = *(const v16h*)(wqkvT + (size_t)(n0 + 16 * ni + mrow) * HID + k0 + hi * 16);
            acc[0][ni] = wmma_f16(a[0].v, b, acc[0][ni]);
            acc[1][ni] = wmma_f16(a[1].v, b, acc[1][ni]);
        }
    }
#pragma unroll
    for (int mi = 0; mi < 2; ++mi)
#pragma unroll
        for (int ni = 0; ni < 4; ++ni)
#pragma unroll
            for (int r = 0; r < 8; ++r) {
                int M = m0 + 16 * mi + (hi ? 8 + r : r);
                qkv[(size_t)M * NQKV + n0 + 16 * ni + mrow] = (_Float16)acc[mi][ni][r];
            }
}

// ---------------------------------------------------------------------------
// Kernel 2: RoPE on Q,K + layout reorg:
//   Qh [NH][SEQ][HD], Kh [NKV][SEQ][HD], Vt [NKV][HD][SEQ]  (all f16)
// ---------------------------------------------------------------------------
__global__ __launch_bounds__(256) void rope_reorg(
    const _Float16* __restrict__ qkv, const int* __restrict__ pos_ids,
    _Float16* __restrict__ Qh, _Float16* __restrict__ Kh, _Float16* __restrict__ Vt)
{
    size_t tid = (size_t)blockIdx.x * 256 + threadIdx.x;  // [0, 2048*3072)
    const size_t QN = (size_t)NH  * SEQ * HD;             // 4194304
    const size_t KN = (size_t)NKV * SEQ * HD;             // 1048576

    if (tid < QN) {                                       // ---- Q + RoPE
        int h = (int)(tid / ((size_t)SEQ * HD));
        int r = (int)(tid % ((size_t)SEQ * HD));
        int q = r / HD, d = r % HD;
        float pos = (float)pos_ids[q];
        float e   = (float)(2 * (d & 31)) / (float)HD;
        float th  = pos * __powf(10000.0f, -e);
        float cv = cosf(th), sv = sinf(th);
        const _Float16* src = qkv + (size_t)q * NQKV + h * HD;
        float xd  = (float)src[d];
        float rot = (d < 32) ? -(float)src[d + 32] : (float)src[d - 32];
        Qh[((size_t)h * SEQ + q) * HD + d] = (_Float16)(xd * cv + rot * sv);
    } else if (tid < QN + KN) {                           // ---- K + RoPE
        size_t t2 = tid - QN;
        int kv = (int)(t2 / ((size_t)SEQ * HD));
        int r  = (int)(t2 % ((size_t)SEQ * HD));
        int q = r / HD, d = r % HD;
        float pos = (float)pos_ids[q];
        float e   = (float)(2 * (d & 31)) / (float)HD;
        float th  = pos * __powf(10000.0f, -e);
        float cv = cosf(th), sv = sinf(th);
        const _Float16* src = qkv + (size_t)q * NQKV + NH * HD + kv * HD;
        float xd  = (float)src[d];
        float rot = (d < 32) ? -(float)src[d + 32] : (float)src[d - 32];
        Kh[((size_t)kv * SEQ + q) * HD + d] = (_Float16)(xd * cv + rot * sv);
    } else {                                              // ---- V transpose
        size_t t2 = tid - QN - KN;
        int kv = (int)(t2 / ((size_t)SEQ * HD));
        int r  = (int)(t2 % ((size_t)SEQ * HD));
        int q = r / HD, d = r % HD;
        Vt[((size_t)kv * HD + d) * SEQ + q] =
            qkv[(size_t)q * NQKV + NH * HD + NKV * HD + kv * HD + d];
    }
}

// ---------------------------------------------------------------------------
// Kernel 3: flash attention.  One wave = one head x 16-query tile.
// Streams keys in chunks of 32; online softmax; 16x64 f32 accumulators.
// P (C-layout) -> A-fragment conversion goes through LDS.
// ---------------------------------------------------------------------------
__global__ __launch_bounds__(128) void attn_kernel(
    const _Float16* __restrict__ Qh, const _Float16* __restrict__ Kh,
    const _Float16* __restrict__ Vt, const float* __restrict__ mask,
    _Float16* __restrict__ attn_out)
{
    __shared__ __align__(32) _Float16 pbuf[4][16][32];   // per-wave P tile

    const int lane = threadIdx.x & 31;
    const int w    = threadIdx.x >> 5;
    const int gw   = blockIdx.x * 4 + w;      // 4096 waves: 32 heads x 128 q-tiles
    const int h    = gw >> 7;
    const int qt   = gw & 127;
    const int kv   = h / GROUPS;
    const int q0   = qt * 16;
    const int mrow = lane & 15, hi = lane >> 4;

    // Q: 16x64 -> two A-fragments (d = 0..31, 32..63)
    V16 aq[2];
    const _Float16* qrow = Qh + ((size_t)h * SEQ + q0 + mrow) * HD;
#pragma unroll
    for (int f = 0; f < 2; ++f) {
        aq[f].h[0] = *(const v8h*)(qrow + f * 32 + hi * 16);
        aq[f].h[1] = *(const v8h*)(qrow + f * 32 + hi * 16 + 8);
    }

    v8f acc[4];
#pragma unroll
    for (int t = 0; t < 4; ++t) acc[t] = zero8();
    float mrun[8], lrun[8];
#pragma unroll
    for (int r = 0; r < 8; ++r) { mrun[r] = -1e30f; lrun[r] = 0.f; }

    const _Float16* kbase = Kh + (size_t)kv * SEQ * HD;
    const _Float16* vbase = Vt + (size_t)kv * HD * SEQ;

    for (int k0 = 0; k0 < SEQ; k0 += 32) {
        // ---- S = Q K^T for 32 keys (two 16x16 tiles, d accumulated over 2 WMMAs)
        v8f s0 = zero8(), s1 = zero8();
#pragma unroll
        for (int f = 0; f < 2; ++f) {
            v16h b0 = *(const v16h*)(kbase + (size_t)(k0      + mrow) * HD + f * 32 + hi * 16);
            v16h b1 = *(const v16h*)(kbase + (size_t)(k0 + 16 + mrow) * HD + f * 32 + hi * 16);
            s0 = wmma_f16(aq[f].v, b0, s0);
            s1 = wmma_f16(aq[f].v, b1, s1);
        }
        float mk0 = mask[k0 + mrow];
        float mk1 = mask[k0 + 16 + mrow];

        // ---- online softmax (rows r / 8+r live in the 16-lane half; xor<=8 stays in-half)
        float alpha[8];
#pragma unroll
        for (int r = 0; r < 8; ++r) {
            float a0 = s0[r] * 0.125f + mk0;
            float a1 = s1[r] * 0.125f + mk1;
            float v = fmaxf(a0, a1);
            v = fmaxf(v, __shfl_xor(v, 1, 32));
            v = fmaxf(v, __shfl_xor(v, 2, 32));
            v = fmaxf(v, __shfl_xor(v, 4, 32));
            v = fmaxf(v, __shfl_xor(v, 8, 32));
            float mnew = fmaxf(mrun[r], v);
            float sc   = __expf(mrun[r] - mnew);
            float p0   = __expf(a0 - mnew);
            float p1   = __expf(a1 - mnew);
            s0[r] = p0; s1[r] = p1;
            float rs = p0 + p1;
            rs += __shfl_xor(rs, 1, 32);
            rs += __shfl_xor(rs, 2, 32);
            rs += __shfl_xor(rs, 4, 32);
            rs += __shfl_xor(rs, 8, 32);
            lrun[r] = lrun[r] * sc + rs;
            mrun[r] = mnew;
            alpha[r] = sc;
        }
#pragma unroll
        for (int t = 0; t < 4; ++t)
#pragma unroll
            for (int r = 0; r < 8; ++r) acc[t][r] = acc[t][r] * alpha[r];

        // ---- P (C-layout) -> LDS -> A-fragment layout
#pragma unroll
        for (int r = 0; r < 8; ++r) {
            int row = hi ? 8 + r : r;
            pbuf[w][row][mrow]      = (_Float16)s0[r];
            pbuf[w][row][16 + mrow] = (_Float16)s1[r];
        }
        __syncthreads();
        V16 pa;
        pa.h[0] = *(const v8h*)&pbuf[w][mrow][hi * 16];
        pa.h[1] = *(const v8h*)&pbuf[w][mrow][hi * 16 + 8];
        __syncthreads();

        // ---- O += P x V   (V transposed: B-fragment is one contiguous v16h)
#pragma unroll
        for (int t = 0; t < 4; ++t) {
            v16h bv = *(const v16h*)(vbase + (size_t)(16 * t + mrow) * SEQ + k0 + hi * 16);
            acc[t] = wmma_f16(pa.v, bv, acc[t]);
        }
    }

    // ---- epilogue: normalize, store f16 [q][h*64+d]
#pragma unroll
    for (int r = 0; r < 8; ++r) {
        float inv = 1.0f / lrun[r];
        int M = hi ? 8 + r : r;
#pragma unroll
        for (int t = 0; t < 4; ++t) {
            attn_out[(size_t)(q0 + M) * (NH * HD) + h * HD + 16 * t + mrow] =
                (_Float16)(acc[t][r] * inv);
        }
    }
}

// ---------------------------------------------------------------------------
// Kernel 4: out = attn(f16) @ woT(f16) -> f32
// Register-blocked 32x64 macro-tile per wave, 8 wmma / K-step.
// ---------------------------------------------------------------------------
#define OUT_NTW (HID / 64)    // 32 macro-tiles along N
__global__ __launch_bounds__(128) void gemm_out(
    const _Float16* __restrict__ attn, const _Float16* __restrict__ woT,
    float* __restrict__ out)
{
    const int lane = threadIdx.x & 31;
    const int w    = threadIdx.x >> 5;
    const int tile = blockIdx.x * 4 + w;          // 64*32 = 2048 macro-tiles
    const int mt   = tile / OUT_NTW;
    const int nt   = tile % OUT_NTW;
    const int m0   = mt * 32, n0 = nt * 64;
    const int mrow = lane & 15, hi = lane >> 4;

    v8f acc[2][4];
#pragma unroll
    for (int mi = 0; mi < 2; ++mi)
#pragma unroll
        for (int ni = 0; ni < 4; ++ni) acc[mi][ni] = zero8();

    const _Float16* ar[2];
    ar[0] = attn + (size_t)(m0 + mrow) * HID;
    ar[1] = attn + (size_t)(m0 + 16 + mrow) * HID;

    for (int k0 = 0; k0 < HID; k0 += 32) {
        __builtin_prefetch(ar[0] + k0 + 64, 0, 0);
        V16 a[2];
#pragma unroll
        for (int mi = 0; mi < 2; ++mi) {
            a[mi].h[0] = *(const v8h*)(ar[mi] + k0 + hi * 16);
            a[mi].h[1] = *(const v8h*)(ar[mi] + k0 + hi * 16 + 8);
        }
#pragma unroll
        for (int ni = 0; ni < 4; ++ni) {
            v16h b = *(const v16h*)(woT + (size_t)(n0 + 16 * ni + mrow) * HID + k0 + hi * 16);
            acc[0][ni] = wmma_f16(a[0].v, b, acc[0][ni]);
            acc[1][ni] = wmma_f16(a[1].v, b, acc[1][ni]);
        }
    }
#pragma unroll
    for (int mi = 0; mi < 2; ++mi)
#pragma unroll
        for (int ni = 0; ni < 4; ++ni)
#pragma unroll
            for (int r = 0; r < 8; ++r) {
                int M = m0 + 16 * mi + (hi ? 8 + r : r);
                out[(size_t)M * HID + n0 + 16 * ni + mrow] = acc[mi][ni][r];
            }
}

// ---------------------------------------------------------------------------
// Host launcher
// ---------------------------------------------------------------------------
extern "C" void kernel_launch(void* const* d_in, const int* in_sizes, int n_in,
                              void* d_out, int out_size, void* d_ws, size_t ws_size,
                              hipStream_t stream) {
    const float* x    = (const float*)d_in[0];
    const float* mask = (const float*)d_in[1];
    const int*   pos  = (const int*)  d_in[2];
    const float* wq   = (const float*)d_in[3];
    const float* wk   = (const float*)d_in[4];
    const float* wv   = (const float*)d_in[5];
    const float* wo   = (const float*)d_in[6];
    float* out = (float*)d_out;

    // workspace carve-out (f16 elements), all 32B-aligned by construction
    _Float16* wqkvT = (_Float16*)d_ws;                               // 3072*2048
    _Float16* woT   = wqkvT + (size_t)NQKV * HID;                    // 2048*2048
    _Float16* qkv   = woT   + (size_t)HID  * HID;                    // 2048*3072
    _Float16* Qh    = qkv   + (size_t)SEQ  * NQKV;                   // 32*2048*64
    _Float16* Kh    = Qh    + (size_t)NH  * SEQ * HD;                //  8*2048*64
    _Float16* Vt    = Kh    + (size_t)NKV * SEQ * HD;                //  8*64*2048
    _Float16* attn  = Vt    + (size_t)NKV * HD * SEQ;                // 2048*2048

    prep_wqkvT<<<(NQKV * HID) / 256, 256, 0, stream>>>(wq, wk, wv, wqkvT);
    prep_woT  <<<(HID  * HID) / 256, 256, 0, stream>>>(wo, woT);
    gemm_qkv  <<<(SEQ / 32) * QKV_NTW / 4, 128, 0, stream>>>(x, wqkvT, qkv);
    rope_reorg<<<(SEQ * NQKV) / 256, 256, 0, stream>>>(qkv, pos, Qh, Kh, Vt);
    attn_kernel<<<(NH * (SEQ / 16)) / 4, 128, 0, stream>>>(Qh, Kh, Vt, mask, attn);
    gemm_out  <<<(SEQ / 32) * OUT_NTW / 4, 128, 0, stream>>>(attn, woT, out);
}
